// RotVQ_61890478735797
// MI455X (gfx1250) — matrix operand order repaired and checked by reading, not compiled, not run on hardware
//
#include <hip/hip_runtime.h>
#include <hip/hip_bf16.h>

typedef __bf16 v16bf __attribute__((ext_vector_type(16)));
typedef __bf16 v8bf  __attribute__((ext_vector_type(8)));
typedef float  v8f   __attribute__((ext_vector_type(8)));

#define Bdim   16
#define Ndim   512
#define Tdim   2048
#define Cdim   1024
#define BT     (Bdim * Tdim)     // 32768 rows of x/tgt
#define KSTEPS (Ndim / 32)       // 16 K-steps of 32 per 16x16 tile

// LDS row pad: 520 halfs -> row stride 1040B = 260 dwords; 260 % 64 = 4,
// so the 16 rows land on 16 distinct 4-dword bank groups (conflict-free b128).
#define TPAD 520

// ---------------------------------------------------------------------------
// Prep: normalize codebook, zero first component, store f32 copy + a bf16
// panel pre-packed in this code's contiguous-k (k = 16h + s) B-matrix (32x16) lane layout:
//   reP[cb][ks][lane][h] = re[cb*16 + lane%16][ks*32 + (lane<16?0:16) + h]
// ---------------------------------------------------------------------------
__global__ void rotvq_prep(const float* __restrict__ rot,
                           float* __restrict__ reF,
                           __bf16* __restrict__ reP) {
  const int c   = blockIdx.x;     // code index 0..1023
  const int tid = threadIdx.x;    // 128 threads
  float v[4];
  float ss = 0.f;
#pragma unroll
  for (int i = 0; i < 4; ++i) {
    v[i] = rot[(size_t)c * Ndim + tid + i * 128];
    ss += v[i] * v[i];
  }
  __shared__ float sred[128];
  sred[tid] = ss;
  __syncthreads();
#pragma unroll
  for (int s = 64; s > 0; s >>= 1) {
    if (tid < s) sred[tid] += sred[tid + s];
    __syncthreads();
  }
  const float inv = 1.0f / sqrtf(sred[0] + 1.1920929e-07f);  // finfo(f32).eps
  const int cb = c >> 4, nn = c & 15;
#pragma unroll
  for (int i = 0; i < 4; ++i) {
    const int k = tid + i * 128;
    const float val = (k == 0) ? 0.0f : v[i] * inv;   // zero col 0, no renorm
    reF[(size_t)c * Ndim + k] = val;
    const int ks = k >> 5, ko = k & 31;
    const int l  = ((ko & 16) ? 16 : 0) + nn;
    const int h  = ko & 15;
    reP[(size_t)(((cb * KSTEPS + ks) * 32 + l) << 4) + h] = (__bf16)val;
  }
}

// ---------------------------------------------------------------------------
// Main: one 16-row tile (16 consecutive t of one b) per 256-thread block.
// Each wave owns 8 code blocks, processed as 4 pairs: per K-step, one pair of
// A loads feeds 4 WMMAs. B panels are register double-buffered to hide L2
// latency. argmin over codes of (tgt@re^T)*(x@re^T); f32 Householder reflect.
// ---------------------------------------------------------------------------
__global__ __launch_bounds__(256)
void rotvq_main(const float* __restrict__ xin, const float* __restrict__ tin,
                const float* __restrict__ reF, const __bf16* __restrict__ reP,
                float* __restrict__ out) {
  __shared__ __align__(16) __bf16 sT[16][TPAD];   // 16.6 KB  tgt (bf16)
  __shared__ __align__(16) __bf16 sX[16][TPAD];   // 16.6 KB  x   (bf16)
  __shared__ float partials[16][17];
  __shared__ float red_val[16][9];
  __shared__ int   red_idx[16][9];
  __shared__ int   sel[16];
  __shared__ float dd[16];

  const int tid  = threadIdx.x;
  const int w    = tid >> 5;
  const int lane = tid & 31;
  const int row0 = blockIdx.x << 4;          // global row = b*T + t
  const int b    = row0 >> 11;               // / T
  const int t0   = row0 & (Tdim - 1);
  const size_t base = (size_t)b * Ndim * Tdim + t0;

  const int r  = tid & 15;   // tile row (t offset)
  const int n0 = tid >> 4;   // feature lane

  // ---- load tile: consecutive lanes -> consecutive t (coalesced) ----
#pragma unroll 4
  for (int k2 = 0; k2 < 32; ++k2) {
    const int n = n0 + (k2 << 4);
    const size_t g = base + (size_t)n * Tdim + r;
    sX[r][n] = (__bf16)xin[g];
    sT[r][n] = (__bf16)tin[g];
  }
  __syncthreads();

  // ---- WMMA phase ----
  const int mrow  = lane & 15;                 // A-matrix M = lane%16
  const int khalf = (lane < 16) ? 0 : 16;      // contiguous K: k = 16h + s (matches reP)
  float minv[8];
  int   mini[8];
#pragma unroll
  for (int j = 0; j < 8; ++j) { minv[j] = 3.4e38f; mini[j] = 0; }

#pragma unroll 1
  for (int cbi = 0; cbi < 4; ++cbi) {
    const int cbA = w + (cbi << 3);            // code blocks w+8k ...
    const int cbB = cbA + 32;                  // ... paired with w+8k+32
    const __bf16* pA = reP + ((size_t)((cbA * KSTEPS) * 32 + lane) << 4);
    const __bf16* pB = reP + ((size_t)((cbB * KSTEPS) * 32 + lane) << 4);
    // prefetch the next pair's first panels into L2/L0
    __builtin_prefetch(pA + ((size_t)(8 * KSTEPS * 32) << 4), 0, 1);
    __builtin_prefetch(pB + ((size_t)(8 * KSTEPS * 32) << 4), 0, 1);

    v8f accTA = {}, accXA = {}, accTB = {}, accXB = {};
    // B register double-buffer: one 32B contiguous load per matrix per step
    v16bf bmA = *(const v16bf*)pA;
    v16bf bmB = *(const v16bf*)pB;
#pragma unroll 4
    for (int ks = 0; ks < KSTEPS; ++ks) {
      const int ksn = (ks + 1) & (KSTEPS - 1);     // wraps on last iter
      const v16bf bnA = *(const v16bf*)(pA + (ksn << 9));
      const v16bf bnB = *(const v16bf*)(pB + (ksn << 9));

      const int kb = (ks << 5) + khalf;
      // A operands: two 16B LDS chunks each, contiguous k = 16h + s
      v8bf t0v = *(const v8bf*)&sT[mrow][kb];
      v8bf t1v = *(const v8bf*)&sT[mrow][kb + 8];
      v16bf aT = __builtin_shufflevector(t0v, t1v,
          0, 1, 2, 3, 4, 5, 6, 7, 8, 9, 10, 11, 12, 13, 14, 15);
      v8bf x0v = *(const v8bf*)&sX[mrow][kb];
      v8bf x1v = *(const v8bf*)&sX[mrow][kb + 8];
      v16bf aX = __builtin_shufflevector(x0v, x1v,
          0, 1, 2, 3, 4, 5, 6, 7, 8, 9, 10, 11, 12, 13, 14, 15);

      accTA = __builtin_amdgcn_wmma_f32_16x16x32_bf16(false, aT, false, bmA,
                                                      (short)0, accTA, false, false);
      accXA = __builtin_amdgcn_wmma_f32_16x16x32_bf16(false, aX, false, bmA,
                                                      (short)0, accXA, false, false);
      accTB = __builtin_amdgcn_wmma_f32_16x16x32_bf16(false, aT, false, bmB,
                                                      (short)0, accTB, false, false);
      accXB = __builtin_amdgcn_wmma_f32_16x16x32_bf16(false, aX, false, bmB,
                                                      (short)0, accXB, false, false);
      bmA = bnA;
      bmB = bnB;
    }
    const int codeA = (cbA << 4) + mrow;       // C-tile: N = lane%16
    const int codeB = (cbB << 4) + mrow;
#pragma unroll
    for (int j = 0; j < 8; ++j) {
      const float pa = accTA[j] * accXA[j];    // ∝ eu_dis up to per-row const
      if (pa < minv[j] || (pa == minv[j] && codeA < mini[j])) {
        minv[j] = pa; mini[j] = codeA;
      }
      const float pb = accTB[j] * accXB[j];
      if (pb < minv[j] || (pb == minv[j] && codeB < mini[j])) {
        minv[j] = pb; mini[j] = codeB;
      }
    }
  }

  // ---- wave-level argmin butterfly over the 16 lanes sharing a row ----
  // C layout: VGPR j holds row M=j in lanes 0-15 and M=j+8 in lanes 16-31.
#pragma unroll
  for (int j = 0; j < 8; ++j) {
    float v2 = minv[j];
    int   i2 = mini[j];
#pragma unroll
    for (int m = 1; m < 16; m <<= 1) {
      const float ov = __shfl_xor(v2, m, 32);
      const int   oi = __shfl_xor(i2, m, 32);
      if (ov < v2 || (ov == v2 && oi < i2)) { v2 = ov; i2 = oi; }
    }
    if (lane == 0)  { red_val[j][w] = v2;     red_idx[j][w] = i2; }
    if (lane == 16) { red_val[j + 8][w] = v2; red_idx[j + 8][w] = i2; }
  }
  __syncthreads();

  // ---- final argmin per row across the 8 waves ----
  if (tid < 16) {
    float bv = 3.4e38f;
    int   bi = 0;
#pragma unroll
    for (int c2 = 0; c2 < 8; ++c2) {
      const float v2 = red_val[tid][c2];
      const int   i2 = red_idx[tid][c2];
      if (v2 < bv || (v2 == bv && i2 < bi)) { bv = v2; bi = i2; }
    }
    sel[tid] = bi;
  }
  __syncthreads();

  // ---- tail: re-read x (f32, coalesced), gather v_sel, reflect, store ----
  float xr[32], vr[32];
  {
    const float* vrow = reF + (size_t)sel[r] * Ndim;
    float s = 0.f;
#pragma unroll 4
    for (int k2 = 0; k2 < 32; ++k2) {
      const int n = n0 + (k2 << 4);
      xr[k2] = xin[base + (size_t)n * Tdim + r];
      vr[k2] = vrow[n];
      s += xr[k2] * vr[k2];
    }
    partials[r][n0] = s;
  }
  __syncthreads();
  if (tid < 16) {
    float s = 0.f;
#pragma unroll
    for (int i = 0; i < 16; ++i) s += partials[tid][i];
    dd[tid] = 2.0f * s;
  }
  __syncthreads();
  {
    const float d2 = dd[r];
#pragma unroll 4
    for (int k2 = 0; k2 < 32; ++k2) {
      const int n = n0 + (k2 << 4);
      out[base + (size_t)n * Tdim + r] = xr[k2] - d2 * vr[k2];
    }
  }
}

// ---------------------------------------------------------------------------
extern "C" void kernel_launch(void* const* d_in, const int* in_sizes, int n_in,
                              void* d_out, int out_size, void* d_ws, size_t ws_size,
                              hipStream_t stream) {
  const float* prev = (const float*)d_in[0];   // [16, 512, 2048]
  const float* tgt  = (const float*)d_in[1];   // [16, 512, 2048]
  const float* rot  = (const float*)d_in[2];   // [1024, 512]
  float* out = (float*)d_out;                  // [16, 512, 2048]

  // workspace: 2 MiB f32 normalized codebook + 1 MiB bf16 WMMA-packed panel
  float*  reF = (float*)d_ws;
  __bf16* reP = (__bf16*)((char*)d_ws + (size_t)Cdim * Ndim * sizeof(float));

  rotvq_prep<<<Cdim, 128, 0, stream>>>(rot, reF, reP);
  rotvq_main<<<BT / 16, 256, 0, stream>>>(prev, tgt, reF, reP, out);
}
